// Decoder_16226386444724
// MI455X (gfx1250) — matrix operand order repaired and checked by reading, not compiled, not run on hardware
//
#include <hip/hip_runtime.h>
#include <hip/hip_bf16.h>
#include <math.h>

typedef __attribute__((ext_vector_type(16))) __bf16 v16bf;
typedef __attribute__((ext_vector_type(8)))  float  v8f;

#define NWG      128
#define TPB      256
#define NTHREADS (NWG * TPB)

// problem sizes
#define Bb    64
#define Ll    512
#define Ee    512
#define Tt    512
#define Ss    512
#define Pp    256
#define Aa    1024
#define Dd    1024
#define Gg    128
#define Vv    148
#define NVp   160      // V padded to a multiple of 16
#define KATT  1792     // P + E + A
#define KDEC  2560     // A + E + D
#define KPROJ 1536     // D + E

#define OUT_ALIGNS ((size_t)Bb * Tt * Vv)
#define OUT_PARAMS (OUT_ALIGNS + (size_t)Bb * Tt * Ll)

struct Params {
  // inputs (f32 unless noted)
  const float* memory;      // (B,L,E)
  const float* dec_in;      // (B,S,T)
  const int*   memlen;      // (B,)
  const float *Wp1, *Wp2;
  const float *WAih, *WAhh, *bAih, *bAhh;
  const float *Wg1f, *bg1, *Wg2f;
  const float *WDih, *WDhh, *bDih, *bDhh;
  const float *Wprojf, *bproj;
  float* out;
  // barrier
  unsigned *barCount, *barGen;
  // workspace
  unsigned short *WATT, *WDEC, *WG1, *WPROJ;   // bf16, fragment-swizzled
  unsigned short *H1, *PRE;                    // bf16 row-major (32768 x 256)
  float *GATT, *GDEC;                          // (64 x 4096)
  unsigned short *AHBF, *DHBF, *CTXBF;         // bf16 states
  float *AC, *DC;                              // f32 cell states
  float *HIDDEN;                               // (64 x 128)
  float *ALPHA;                                // (64 x 512)
  float *MEAN;                                 // (64)
};

__device__ __forceinline__ unsigned short f2bf(float f) {
  unsigned u = __float_as_uint(f);
  unsigned r = u + 0x7FFFu + ((u >> 16) & 1u);
  return (unsigned short)(r >> 16);
}
__device__ __forceinline__ float bf2f(unsigned short b) {
  return __uint_as_float(((unsigned)b) << 16);
}
__device__ __forceinline__ float sigm(float x) { return 1.0f / (1.0f + expf(-x)); }

__device__ __forceinline__ float dropmask(unsigned idx, unsigned seed) {
  unsigned x = idx ^ seed;
  x *= 0x9E3779B1u; x ^= x >> 16; x *= 0x85EBCA6Bu; x ^= x >> 13;
  x *= 0xC2B2AE35u; x ^= x >> 16;
  return (x & 1u) ? 2.0f : 0.0f;   // p=0.5 dropout, 1/p scaling
}

// grid-wide sense-reversing barrier (all 128 WGs are resident)
__device__ __forceinline__ void gbar(unsigned* cnt, unsigned* gen) {
  __syncthreads();
  if (threadIdx.x == 0) {
    __threadfence();
    unsigned g = __hip_atomic_load(gen, __ATOMIC_RELAXED, __HIP_MEMORY_SCOPE_AGENT);
    unsigned a = __hip_atomic_fetch_add(cnt, 1u, __ATOMIC_ACQ_REL, __HIP_MEMORY_SCOPE_AGENT);
    if (a == NWG - 1) {
      __hip_atomic_store(cnt, 0u, __ATOMIC_RELAXED, __HIP_MEMORY_SCOPE_AGENT);
      __hip_atomic_fetch_add(gen, 1u, __ATOMIC_RELEASE, __HIP_MEMORY_SCOPE_AGENT);
    } else {
      while (__hip_atomic_load(gen, __ATOMIC_ACQUIRE, __HIP_MEMORY_SCOPE_AGENT) == g) {
        __builtin_amdgcn_s_sleep(2);
      }
    }
  }
  __syncthreads();
}

// A fragment: 16x32 bf16, row-major source. lane<16: K=k0+[0..7],[16..23]; lane>=16: +8
__device__ __forceinline__ v16bf load_a(const unsigned short* base, int ld, int row0, int k0) {
  int lane = threadIdx.x & 31;
  const unsigned short* p = base + (size_t)(row0 + (lane & 15)) * ld + k0 + ((lane >> 4) << 3);
  v16bf r;
  *(uint4*)&r       = *(const uint4*)p;        // K offsets +0..7
  *((uint4*)&r + 1) = *(const uint4*)(p + 16); // K offsets +16..23
  return r;
}
// B fragment from pre-swizzled weights: one contiguous 32B chunk per lane
__device__ __forceinline__ v16bf load_b(const unsigned short* W, int KT, int kt, int nt) {
  int lane = threadIdx.x & 31;
  const unsigned short* p = W + (((size_t)(nt * KT + kt) * 32 + lane) << 4);
  v16bf r;
  *(uint4*)&r       = *(const uint4*)p;
  *((uint4*)&r + 1) = *(const uint4*)(p + 8);
  return r;
}
__device__ __forceinline__ v8f wmma_bf16(v16bf a, v16bf b, v8f c) {
  return __builtin_amdgcn_wmma_f32_16x16x32_bf16(false, a, false, b, (short)0, c, false, false);
}

// pack f32 source into bf16 B-fragment-swizzled layout
template <typename F>
__device__ __forceinline__ void packB(unsigned short* dst, int K, int N, int gtid, F src) {
  const int KT = K >> 5;
  const size_t total = (size_t)K * N;
  for (size_t idx = gtid; idx < total; idx += NTHREADS) {
    int e    = (int)(idx & 15);
    int lane = (int)((idx >> 4) & 31);
    size_t tile = idx >> 9;
    int kt = (int)(tile % KT);
    int nt = (int)(tile / KT);
    int k = (kt << 5) + ((lane >> 4) << 3) + (e & 7) + ((e >> 3) << 4);
    int n = (nt << 4) + (lane & 15);
    dst[idx] = f2bf(src(k, n));
  }
}

__global__ __launch_bounds__(TPB, 1) void gmm_decoder_persistent(Params p) {
  const int tid  = threadIdx.x;
  const int wg   = blockIdx.x;
  const int gtid = wg * TPB + tid;
  const int wid  = gtid >> 5;           // 0..1023
  const int lane = tid & 31;

  __shared__ float smem[8 * 512];       // 16 KB, reused across phases

  // ---------------- P0: pack weights to bf16 (swizzled), init state ----------------
  packB(p.WATT, KATT, 4096, gtid, [&](int k, int n) -> float {
    return (k < Pp + Ee) ? p.WAih[(size_t)k * 4096 + n] : p.WAhh[(size_t)(k - (Pp + Ee)) * 4096 + n];
  });
  packB(p.WDEC, KDEC, 4096, gtid, [&](int k, int n) -> float {
    return (k < Aa + Ee) ? p.WDih[(size_t)k * 4096 + n] : p.WDhh[(size_t)(k - (Aa + Ee)) * 4096 + n];
  });
  packB(p.WG1, Aa, Gg, gtid, [&](int k, int n) -> float { return p.Wg1f[(size_t)k * Gg + n]; });
  packB(p.WPROJ, KPROJ, NVp, gtid, [&](int k, int n) -> float {
    return (n < Vv) ? p.Wprojf[(size_t)k * Vv + n] : 0.0f;
  });
  for (int i = gtid; i < Bb * Aa; i += NTHREADS) { p.AC[i] = 0.0f; p.DC[i] = 0.0f; p.AHBF[i] = 0; p.DHBF[i] = 0; }
  for (int i = gtid; i < Bb * Ee; i += NTHREADS) p.CTXBF[i] = 0;
  for (int i = gtid; i < Bb; i += NTHREADS) p.MEAN[i] = 0.0f;
  gbar(p.barCount, p.barGen);

  // ---------------- P1: prenet layer 1 (x @ Wp1, relu, dropout) -> H1 bf16 ----------
  for (int rowBase = wg * 8; rowBase < Tt * Bb; rowBase += NWG * 8) {
    __syncthreads();
    for (int i = tid; i < 8 * 512; i += TPB) {
      int r = i >> 9, s = i & 511;
      int m = rowBase + r;
      int tt = m >> 6, b = m & 63;
      smem[i] = (tt == 0) ? 0.0f : p.dec_in[((size_t)b * Ss + s) * Tt + (tt - 1)];
    }
    __syncthreads();
    float acc[8] = {};
    for (int s = 0; s < Ss; ++s) {
      float w = p.Wp1[(size_t)s * Pp + tid];
      #pragma unroll
      for (int r = 0; r < 8; ++r) acc[r] = fmaf(smem[(r << 9) + s], w, acc[r]);
    }
    #pragma unroll
    for (int r = 0; r < 8; ++r) {
      int m = rowBase + r;
      float h = fmaxf(acc[r], 0.0f) * dropmask((unsigned)(m * Pp + tid), 0x12345u);
      p.H1[(size_t)m * Pp + tid] = f2bf(h);
    }
  }
  gbar(p.barCount, p.barGen);

  // ---------------- P2: prenet layer 2 (H1 @ Wp2, relu, dropout) -> PRE bf16 --------
  for (int rowBase = wg * 8; rowBase < Tt * Bb; rowBase += NWG * 8) {
    __syncthreads();
    for (int i = tid; i < 8 * 256; i += TPB) {
      int r = i >> 8, s = i & 255;
      smem[i] = bf2f(p.H1[(size_t)(rowBase + r) * Pp + s]);
    }
    __syncthreads();
    float acc[8] = {};
    for (int s = 0; s < Pp; ++s) {
      float w = p.Wp2[(size_t)s * Pp + tid];
      #pragma unroll
      for (int r = 0; r < 8; ++r) acc[r] = fmaf(smem[(r << 8) + s], w, acc[r]);
    }
    #pragma unroll
    for (int r = 0; r < 8; ++r) {
      int m = rowBase + r;
      float h = fmaxf(acc[r], 0.0f) * dropmask((unsigned)(m * Pp + tid), 0xABCDEu);
      p.PRE[(size_t)m * Pp + tid] = f2bf(h);
    }
  }
  gbar(p.barCount, p.barGen);

  // ---------------- sequential decoder: 512 steps, 7 global barriers each ----------
  for (int t = 0; t < Tt; ++t) {
    // Phase A: attention gates = [pre_t | ctx | ah] @ WATT   (1024 tiles, 1/wave)
    {
      const int KT = KATT >> 5;
      int mt = wid & 3, nt = wid >> 2;
      v8f acc0 = {}, acc1 = {};
      for (int kt = 0; kt < KT; kt += 2) {
        #pragma unroll
        for (int u = 0; u < 2; ++u) {
          int k0 = (kt + u) << 5;
          const unsigned short* base; int ld, koff;
          if (k0 < Pp)            { base = p.PRE + (size_t)t * Bb * Pp; ld = Pp;  koff = k0; }
          else if (k0 < Pp + Ee)  { base = p.CTXBF;                     ld = Ee;  koff = k0 - Pp; }
          else                    { base = p.AHBF;                      ld = Aa;  koff = k0 - (Pp + Ee); }
          v16bf a = load_a(base, ld, mt * 16, koff);
          v16bf b = load_b(p.WATT, KT, kt + u, nt);
          if (u == 0) acc0 = wmma_bf16(a, b, acc0); else acc1 = wmma_bf16(a, b, acc1);
        }
      }
      v8f acc = acc0 + acc1;
      int m0 = mt * 16 + ((lane >> 4) << 3), n = nt * 16 + (lane & 15);
      #pragma unroll
      for (int r = 0; r < 8; ++r) p.GATT[(size_t)(m0 + r) * 4096 + n] = acc[r];
    }
    gbar(p.barCount, p.barGen);

    // Phase B: attention LSTM nonlinearity -> AC, AHBF
    for (int idx = gtid; idx < Bb * Aa; idx += NTHREADS) {
      int b = idx >> 10, n = idx & 1023;
      const float* Gr = p.GATT + (size_t)b * 4096;
      float gi = Gr[n]          + p.bAih[n]          + p.bAhh[n];
      float gf = Gr[n + 1024]   + p.bAih[n + 1024]   + p.bAhh[n + 1024];
      float gg = Gr[n + 2048]   + p.bAih[n + 2048]   + p.bAhh[n + 2048];
      float go = Gr[n + 3072]   + p.bAih[n + 3072]   + p.bAhh[n + 3072];
      float c  = sigm(gf) * p.AC[idx] + sigm(gi) * tanhf(gg);
      p.AC[idx]   = c;
      p.AHBF[idx] = f2bf(sigm(go) * tanhf(c));
    }
    gbar(p.barCount, p.barGen);

    // Phase C1: hidden = tanh(ah @ Wg1 + bg1)   (32 tiles)
    if (wid < 32) {
      const int KT = Aa >> 5;
      int mt = wid & 3, nt = wid >> 2;
      v8f acc0 = {}, acc1 = {};
      for (int kt = 0; kt < KT; kt += 2) {
        v16bf a0 = load_a(p.AHBF, Aa, mt * 16, kt << 5);
        v16bf b0 = load_b(p.WG1, KT, kt, nt);
        acc0 = wmma_bf16(a0, b0, acc0);
        v16bf a1 = load_a(p.AHBF, Aa, mt * 16, (kt + 1) << 5);
        v16bf b1 = load_b(p.WG1, KT, kt + 1, nt);
        acc1 = wmma_bf16(a1, b1, acc1);
      }
      v8f acc = acc0 + acc1;
      int m0 = mt * 16 + ((lane >> 4) << 3), n = nt * 16 + (lane & 15);
      #pragma unroll
      for (int r = 0; r < 8; ++r) p.HIDDEN[(size_t)(m0 + r) * Gg + n] = tanhf(acc[r] + p.bg1[n]);
    }
    gbar(p.barCount, p.barGen);

    // Phase C2: per-batch GMM params + alpha (one WG per batch row)
    if (wg < Bb) {
      float pm0 = 0.0f, pm1 = 0.0f;
      if (tid < Gg) {
        float v = p.HIDDEN[(size_t)wg * Gg + tid];
        pm0 = v * p.Wg2f[tid * 2 + 0];
        pm1 = v * p.Wg2f[tid * 2 + 1];
      }
      smem[tid] = pm0; smem[256 + tid] = pm1;
      __syncthreads();
      for (int s2 = 64; s2 >= 1; s2 >>= 1) {
        if (tid < s2) { smem[tid] += smem[tid + s2]; smem[256 + tid] += smem[256 + tid + s2]; }
        __syncthreads();
      }
      if (tid == 0) {
        float hm = smem[0], hs = smem[256];
        float mean  = p.MEAN[wg] + expf(hm) * 8.0f;
        float scale = expf(hs) * 8.0f;
        p.MEAN[wg] = mean;
        p.out[OUT_PARAMS + ((size_t)wg * Tt + t) * 2 + 0] = hm;
        p.out[OUT_PARAMS + ((size_t)wg * Tt + t) * 2 + 1] = hs;
        smem[0] = mean; smem[1] = scale;
      }
      __syncthreads();
      float mean = smem[0], scale = smem[1];
      float inv2s2 = 0.5f / (scale * scale);
      float Zinv   = 1.0f / sqrtf(6.283185307179586f * scale * scale);
      int ml = p.memlen[wg];
      for (int l = tid; l < Ll; l += TPB) {
        float d = (float)l - mean;
        float a = (l < ml) ? expf(-d * d * inv2s2) * Zinv : 0.0f;
        p.ALPHA[(size_t)wg * Ll + l] = a;
        p.out[OUT_ALIGNS + ((size_t)wg * Tt + t) * Ll + l] = a;
      }
    }
    gbar(p.barCount, p.barGen);

    // Phase C3: ctx[b,e] = sum_l alpha[b,l] * memory[b,l,e]  (one thread per output)
    for (int idx = gtid; idx < Bb * Ee; idx += NTHREADS) {
      int b = idx >> 9, e = idx & 511;
      const float* mem = p.memory + ((size_t)b * Ll) * Ee + e;
      const float* al  = p.ALPHA + (size_t)b * Ll;
      float acc = 0.0f;
      #pragma unroll 8
      for (int l = 0; l < Ll; ++l) acc = fmaf(al[l], mem[(size_t)l * Ee], acc);
      p.CTXBF[idx] = f2bf(acc);
    }
    gbar(p.barCount, p.barGen);

    // Phase D: decoder gates = [ah | ctx | dh] @ WDEC   (1024 tiles)
    {
      const int KT = KDEC >> 5;
      int mt = wid & 3, nt = wid >> 2;
      v8f acc0 = {}, acc1 = {};
      for (int kt = 0; kt < KT; kt += 2) {
        #pragma unroll
        for (int u = 0; u < 2; ++u) {
          int k0 = (kt + u) << 5;
          const unsigned short* base; int ld, koff;
          if (k0 < Aa)            { base = p.AHBF;  ld = Aa; koff = k0; }
          else if (k0 < Aa + Ee)  { base = p.CTXBF; ld = Ee; koff = k0 - Aa; }
          else                    { base = p.DHBF;  ld = Dd; koff = k0 - (Aa + Ee); }
          v16bf a = load_a(base, ld, mt * 16, koff);
          v16bf b = load_b(p.WDEC, KT, kt + u, nt);
          if (u == 0) acc0 = wmma_bf16(a, b, acc0); else acc1 = wmma_bf16(a, b, acc1);
        }
      }
      v8f acc = acc0 + acc1;
      int m0 = mt * 16 + ((lane >> 4) << 3), n = nt * 16 + (lane & 15);
      #pragma unroll
      for (int r = 0; r < 8; ++r) p.GDEC[(size_t)(m0 + r) * 4096 + n] = acc[r];
    }
    gbar(p.barCount, p.barGen);

    // Phase E: decoder LSTM nonlinearity -> DC, DHBF
    for (int idx = gtid; idx < Bb * Dd; idx += NTHREADS) {
      int b = idx >> 10, n = idx & 1023;
      const float* Gr = p.GDEC + (size_t)b * 4096;
      float gi = Gr[n]        + p.bDih[n]        + p.bDhh[n];
      float gf = Gr[n + 1024] + p.bDih[n + 1024] + p.bDhh[n + 1024];
      float gg = Gr[n + 2048] + p.bDih[n + 2048] + p.bDhh[n + 2048];
      float go = Gr[n + 3072] + p.bDih[n + 3072] + p.bDhh[n + 3072];
      float c  = sigm(gf) * p.DC[idx] + sigm(gi) * tanhf(gg);
      p.DC[idx]   = c;
      p.DHBF[idx] = f2bf(sigm(go) * tanhf(c));
    }
    gbar(p.barCount, p.barGen);

    // Phase F: logits = [dh | ctx] @ WPROJ + bproj  (40 tiles; no trailing barrier,
    // overlaps with next step's Phase A — no data hazard across that boundary)
    if (wid < 40) {
      const int KT = KPROJ >> 5;
      int mt = wid & 3, nt = wid >> 2;   // nt 0..9
      v8f acc0 = {}, acc1 = {};
      for (int kt = 0; kt < KT; kt += 2) {
        #pragma unroll
        for (int u = 0; u < 2; ++u) {
          int k0 = (kt + u) << 5;
          const unsigned short* base; int ld, koff;
          if (k0 < Dd) { base = p.DHBF;  ld = Dd; koff = k0; }
          else         { base = p.CTXBF; ld = Ee; koff = k0 - Dd; }
          v16bf a = load_a(base, ld, mt * 16, koff);
          v16bf b = load_b(p.WPROJ, KT, kt + u, nt);
          if (u == 0) acc0 = wmma_bf16(a, b, acc0); else acc1 = wmma_bf16(a, b, acc1);
        }
      }
      v8f acc = acc0 + acc1;
      int m0 = mt * 16 + ((lane >> 4) << 3), n = nt * 16 + (lane & 15);
      if (n < Vv) {
        #pragma unroll
        for (int r = 0; r < 8; ++r)
          p.out[((size_t)(m0 + r) * Tt + t) * Vv + n] = acc[r] + p.bproj[n];
      }
    }
    // no barrier: next Phase A's reads (PRE/CTXBF/AHBF) are already fenced by
    // this step's earlier barriers, and Phase B's Gatt reads sit behind A->B barrier.
  }
}

extern "C" void kernel_launch(void* const* d_in, const int* in_sizes, int n_in,
                              void* d_out, int out_size, void* d_ws, size_t ws_size,
                              hipStream_t stream) {
  (void)in_sizes; (void)n_in; (void)out_size;

  char* ws = (char*)d_ws;
  size_t off = 0;
  auto carve = [&](size_t bytes) -> void* {
    void* q = ws + off;
    off += (bytes + 255) & ~(size_t)255;
    return q;
  };

  unsigned* bar          = (unsigned*)carve(256);
  unsigned short* WATT   = (unsigned short*)carve((size_t)KATT * 4096 * 2);
  unsigned short* WDEC   = (unsigned short*)carve((size_t)KDEC * 4096 * 2);
  unsigned short* WG1    = (unsigned short*)carve((size_t)Aa * Gg * 2);
  unsigned short* WPROJ  = (unsigned short*)carve((size_t)KPROJ * NVp * 2);
  unsigned short* H1     = (unsigned short*)carve((size_t)Tt * Bb * Pp * 2);
  unsigned short* PRE    = (unsigned short*)carve((size_t)Tt * Bb * Pp * 2);
  float* GATT            = (float*)carve((size_t)Bb * 4096 * 4);
  float* GDEC            = (float*)carve((size_t)Bb * 4096 * 4);
  unsigned short* AHBF   = (unsigned short*)carve((size_t)Bb * Aa * 2);
  unsigned short* DHBF   = (unsigned short*)carve((size_t)Bb * Dd * 2);
  unsigned short* CTXBF  = (unsigned short*)carve((size_t)Bb * Ee * 2);
  float* AC              = (float*)carve((size_t)Bb * Aa * 4);
  float* DC              = (float*)carve((size_t)Bb * Dd * 4);
  float* HIDDEN          = (float*)carve((size_t)Bb * Gg * 4);
  float* ALPHA           = (float*)carve((size_t)Bb * Ll * 4);
  float* MEAN            = (float*)carve((size_t)Bb * 4);

  if (off > ws_size) return;  // workspace too small; nothing safe to do

  hipMemsetAsync((void*)bar, 0, 256, stream);  // stream op: graph-capture legal

  Params p;
  p.memory = (const float*)d_in[0];
  p.dec_in = (const float*)d_in[1];
  p.memlen = (const int*)d_in[2];
  p.Wp1    = (const float*)d_in[3];
  p.Wp2    = (const float*)d_in[4];
  p.WAih   = (const float*)d_in[5];
  p.WAhh   = (const float*)d_in[6];
  p.bAih   = (const float*)d_in[7];
  p.bAhh   = (const float*)d_in[8];
  p.Wg1f   = (const float*)d_in[9];
  p.bg1    = (const float*)d_in[10];
  p.Wg2f   = (const float*)d_in[11];
  p.WDih   = (const float*)d_in[12];
  p.WDhh   = (const float*)d_in[13];
  p.bDih   = (const float*)d_in[14];
  p.bDhh   = (const float*)d_in[15];
  p.Wprojf = (const float*)d_in[16];
  p.bproj  = (const float*)d_in[17];
  p.out    = (float*)d_out;
  p.barCount = bar;
  p.barGen   = bar + 16;
  p.WATT = WATT; p.WDEC = WDEC; p.WG1 = WG1; p.WPROJ = WPROJ;
  p.H1 = H1; p.PRE = PRE;
  p.GATT = GATT; p.GDEC = GDEC;
  p.AHBF = AHBF; p.DHBF = DHBF; p.CTXBF = CTXBF;
  p.AC = AC; p.DC = DC;
  p.HIDDEN = HIDDEN; p.ALPHA = ALPHA; p.MEAN = MEAN;

  gmm_decoder_persistent<<<NWG, TPB, 0, stream>>>(p);
}
